// GAT_60129542144781
// MI455X (gfx1250) — matrix operand order repaired and checked by reading, not compiled, not run on hardware
//
#include <hip/hip_runtime.h>
#include <hip/hip_bf16.h>

#define NN 4096
#define NWORDS (NN / 32)
#define SPLITJ 8
#define SLOPE 0.01f
#define NEG_BIG (-1e9f)
#define NEG_INF (-3.0e38f)

typedef __attribute__((ext_vector_type(16))) _Float16 v16h;
typedef __attribute__((ext_vector_type(8)))  float    v8f;

// ---------------- adj int32 [N,N] -> bitmask [N, N/32] ----------------
__global__ void pack_mask(const int* __restrict__ adj, unsigned* __restrict__ bits) {
    int w = blockIdx.x * blockDim.x + threadIdx.x;
    if (w >= NN * NWORDS) return;
    const int* a = adj + (size_t)w * 32;
    unsigned m = 0u;
    #pragma unroll
    for (int b = 0; b < 32; ++b) m |= (a[b] > 0 ? 1u : 0u) << b;
    bits[w] = m;
}

// ---------------- h = x @ W ; s = <h,a_src>, t = <h,a_dst> per head ----------------
// Also emits h as f16, transposed + zero-padded to 16 columns: hpadT[col][n].
// grid: NN blocks, block: 32 threads (HD <= 16, H <= 3)
__global__ __launch_bounds__(32)
void gat_transform(const float* __restrict__ x, const float* __restrict__ W,
                   const float* __restrict__ a_src, const float* __restrict__ a_dst,
                   _Float16* __restrict__ hpadT, float* __restrict__ s, float* __restrict__ t,
                   int F, int H, int D) {
    const int n = blockIdx.x;
    const int HD = H * D;
    __shared__ float hloc[16];
    const int o = threadIdx.x;
    float acc = 0.f;
    if (o < HD) {
        const float* xr = x + (size_t)n * F;
        for (int f = 0; f < F; ++f) acc = fmaf(xr[f], W[f * HD + o], acc);
        hloc[o] = acc;
    }
    if (o < 16) hpadT[(size_t)o * NN + n] = (o < HD) ? (_Float16)acc : (_Float16)0.f;
    __syncthreads();
    if (o < H) {
        float ss = 0.f, tt = 0.f;
        for (int d = 0; d < D; ++d) {
            float hv = hloc[o * D + d];
            ss = fmaf(hv, a_src[o * D + d], ss);
            tt = fmaf(hv, a_dst[o * D + d], tt);
        }
        s[n * H + o] = ss;
        t[n * H + o] = tt;
    }
}

// ---------------- per-row score max (no exp: LReLU is monotonic) ----------------
// mrow[i,h] = anyMasked ? LReLU(s_i + max_{j in mask} t_j) : -1e9
template<int H>
__global__ __launch_bounds__(256)
void gat_rowmax(const unsigned* __restrict__ bits, const float* __restrict__ s,
                const float* __restrict__ t, float* __restrict__ mrow) {
    const int i = blockIdx.x;
    const int tid = threadIdx.x;
    float tmax[H];
    #pragma unroll
    for (int hh = 0; hh < H; ++hh) tmax[hh] = NEG_INF;
    const unsigned* rb = bits + (size_t)i * NWORDS;
    for (int j = tid; j < NN; j += 256) {
        const unsigned w = rb[j >> 5];
        const bool on = (w >> (j & 31)) & 1u;
        #pragma unroll
        for (int hh = 0; hh < H; ++hh) {
            float tv = on ? t[j * H + hh] : NEG_INF;
            tmax[hh] = fmaxf(tmax[hh], tv);
        }
    }
    __shared__ float lm[256];
    #pragma unroll
    for (int hh = 0; hh < H; ++hh) {
        lm[tid] = tmax[hh];
        __syncthreads();
        for (int st = 128; st > 0; st >>= 1) {
            if (tid < st) lm[tid] = fmaxf(lm[tid], lm[tid + st]);
            __syncthreads();
        }
        if (tid == 0) {
            float tm = lm[0];
            float mr;
            if (tm < -1.0e38f) mr = NEG_BIG;                 // row has no neighbors
            else { float e = s[i * H + hh] + tm; mr = e > 0.f ? e : SLOPE * e; }
            mrow[i * H + hh] = mr;
        }
        __syncthreads();
    }
}

// ---------------- WMMA attention aggregation ----------------
// grid: (NN/16, H, SPLITJ), block: 32 (one wave).
// A = unnormalized p tile 16x32 f16 (computed in-register, t staged to LDS via
// gfx1250 async global->LDS copies), B = hpadT tile 32x16 f16 via 2x b128 loads.
// Also accumulates per-row denominator l = sum(p); normalization happens in combine.
template<int H, int D>
__global__ __launch_bounds__(32)
void gat_agg(const unsigned* __restrict__ bits, const _Float16* __restrict__ hpadT,
             const float* __restrict__ sbuf, const float* __restrict__ tbuf,
             const float* __restrict__ mbuf,
             float* __restrict__ cpart, float* __restrict__ lpart) {
    const int i0   = blockIdx.x << 4;
    const int hh   = blockIdx.y;
    const int z    = blockIdx.z;
    const int lane = threadIdx.x;
    const int HD   = H * D;
    const int m    = lane & 15;   // A row / B,C column
    const int g    = lane >> 4;   // lane-half select
    const int row  = i0 + m;
    const float srow = sbuf[row * H + hh];
    const float mr   = mbuf[row * H + hh];
    const unsigned* rb = bits + (size_t)row * NWORDS;
    const int JLEN = NN / SPLITJ;
    const int jbeg = z * JLEN;

    // ---- stage t[jbeg .. jbeg+JLEN) for this head into LDS (async copy path) ----
    __shared__ float tlds[NN / SPLITJ];
    {
        const unsigned lbase = (unsigned)(size_t)(&tlds[0]);
        #pragma unroll
        for (int i = 0; i < (NN / SPLITJ) / 32; ++i) {
            const int jl = lane + 32 * i;
            const float* ga = tbuf + (size_t)(jbeg + jl) * H + hh;
            const unsigned la = lbase + (unsigned)(jl * 4);
            asm volatile("global_load_async_to_lds_b32 %0, %1, off"
                         :: "v"(la), "v"(ga) : "memory");
        }
        asm volatile("s_wait_asynccnt 0x0" ::: "memory");
    }

    // B base: column cm of hpadT; pad lanes (m >= D) point at a zero column (<16 since HD<16 there)
    const int cm = (m < D) ? (hh * D + m) : 15;
    const _Float16* bp = hpadT + (size_t)cm * NN + (g << 3);

    v8f c = {};
    float psum = 0.f;
    for (int j0 = jbeg; j0 < jbeg + JLEN; j0 += 32) {
        const unsigned w = rb[j0 >> 5];
        // t values for this lane's K pattern: two contiguous runs of 8 -> 4x ds_load_b128
        const float* tp = &tlds[(j0 - jbeg) + (g << 3)];
        const float4 t0 = *(const float4*)(tp);
        const float4 t1 = *(const float4*)(tp + 4);
        const float4 t2 = *(const float4*)(tp + 16);
        const float4 t3 = *(const float4*)(tp + 20);
        const float tv[16] = {t0.x, t0.y, t0.z, t0.w, t1.x, t1.y, t1.z, t1.w,
                              t2.x, t2.y, t2.z, t2.w, t3.x, t3.y, t3.z, t3.w};
        v16h a;
        #pragma unroll
        for (int q = 0; q < 16; ++q) {   // ISA A-layout: q<8 -> k=8g+q ; q>=8 -> k=8g+16+(q-8)
            const int k = (q < 8) ? ((g << 3) + q) : ((g << 3) + 8 + q);
            float e = srow + tv[q];
            e = e > 0.f ? e : SLOPE * e;
            e = ((w >> k) & 1u) ? e : NEG_BIG;
            const _Float16 ph = (_Float16)__expf(e - mr);   // p in [0,1]
            a[q] = ph;
            psum += (float)ph;                              // denominator from the f16-rounded p
        }
        const v16h b = __builtin_shufflevector(*(const __attribute__((ext_vector_type(8))) _Float16*)(bp + j0), *(const __attribute__((ext_vector_type(8))) _Float16*)(bp + j0 + 16), 0, 1, 2, 3, 4, 5, 6, 7, 8, 9, 10, 11, 12, 13, 14, 15);             // two global_load_b128
        c = __builtin_amdgcn_wmma_f32_16x16x32_f16(false, a, false, b, (short)0, c,
                                                   false, false);
    }

    // per-row denominator: lanes m and m+16 cover complementary halves of each j-tile
    psum += __shfl_xor(psum, 16, 32);
    if (lane < 16) lpart[(size_t)z * (NN * H) + row * H + hh] = psum;

    float* dst = cpart + (size_t)z * ((size_t)NN * HD);
    if (m < D) {
        #pragma unroll
        for (int r = 0; r < 8; ++r) {   // C: lane holds col m, rows r + 8*g
            const int orow = i0 + r + (g << 3);
            dst[(size_t)orow * HD + hh * D + m] = c[r];
        }
    }
}

// Fixed-order combine of j-split partials + softmax normalization (deterministic)
__global__ void combine_norm(const float* __restrict__ cpart, const float* __restrict__ lpart,
                             float* __restrict__ out, int HD, int H, int D) {
    int idx = blockIdx.x * blockDim.x + threadIdx.x;
    if (idx >= NN * HD) return;
    const int i = idx / HD;
    const int hh = (idx % HD) / D;
    float sp = 0.f, sl = 0.f;
    for (int zz = 0; zz < SPLITJ; ++zz) {
        sp += cpart[(size_t)zz * (NN * HD) + idx];
        sl += lpart[(size_t)zz * (NN * H) + i * H + hh];
    }
    out[idx] = sp / sl;
}

// ---------------- ELU + LayerNorm (+ optional write into LSTM sequence slot) ----------------
template<int HD>
__global__ void elu_ln(const float* __restrict__ in, const float* __restrict__ gw,
                       const float* __restrict__ bw, float* __restrict__ outp,
                       float* __restrict__ seq_slot, int seq_stride) {
    int n = blockIdx.x * blockDim.x + threadIdx.x;
    if (n >= NN) return;
    float v[HD];
    float mu = 0.f;
    #pragma unroll
    for (int f = 0; f < HD; ++f) {
        float x = in[(size_t)n * HD + f];
        x = x > 0.f ? x : (__expf(x) - 1.f);     // ELU
        v[f] = x; mu += x;
    }
    mu *= (1.f / HD);
    float var = 0.f;
    #pragma unroll
    for (int f = 0; f < HD; ++f) { float d = v[f] - mu; var += d * d; }
    var *= (1.f / HD);
    float rs = rsqrtf(var + 1e-5f);
    #pragma unroll
    for (int f = 0; f < HD; ++f) {
        float y = (v[f] - mu) * rs * gw[f] + bw[f];
        outp[(size_t)n * HD + f] = y;
        if (seq_slot) seq_slot[(size_t)n * seq_stride + f] = y;
    }
}

// ---------------- per-node LSTM over T=6 steps, torch gate order i,f,g,o ----------------
template<int H>
__global__ void lstm_kernel(const float* __restrict__ seq, const float* __restrict__ Wih,
                            const float* __restrict__ Whh, const float* __restrict__ bih,
                            const float* __restrict__ bhh, float* __restrict__ hn, int T) {
    int n = blockIdx.x * blockDim.x + threadIdx.x;
    if (n >= NN) return;
    float h[H], c[H];
    #pragma unroll
    for (int k = 0; k < H; ++k) { h[k] = 0.f; c[k] = 0.f; }
    for (int tt = 0; tt < T; ++tt) {
        const float* xt = seq + ((size_t)n * T + tt) * H;
        float xv[H];
        #pragma unroll
        for (int k = 0; k < H; ++k) xv[k] = xt[k];
        float gi[H], gf[H], gg[H], go[H];
        #pragma unroll
        for (int r = 0; r < H; ++r) {
            float a0 = bih[r]         + bhh[r];
            float a1 = bih[H + r]     + bhh[H + r];
            float a2 = bih[2 * H + r] + bhh[2 * H + r];
            float a3 = bih[3 * H + r] + bhh[3 * H + r];
            #pragma unroll
            for (int k = 0; k < H; ++k) {
                a0 = fmaf(xv[k], Wih[r * H + k], a0);            a0 = fmaf(h[k], Whh[r * H + k], a0);
                a1 = fmaf(xv[k], Wih[(H + r) * H + k], a1);      a1 = fmaf(h[k], Whh[(H + r) * H + k], a1);
                a2 = fmaf(xv[k], Wih[(2 * H + r) * H + k], a2);  a2 = fmaf(h[k], Whh[(2 * H + r) * H + k], a2);
                a3 = fmaf(xv[k], Wih[(3 * H + r) * H + k], a3);  a3 = fmaf(h[k], Whh[(3 * H + r) * H + k], a3);
            }
            gi[r] = a0; gf[r] = a1; gg[r] = a2; go[r] = a3;
        }
        #pragma unroll
        for (int r = 0; r < H; ++r) {
            float ir = 1.f / (1.f + __expf(-gi[r]));
            float fr = 1.f / (1.f + __expf(-gf[r]));
            float gr = tanhf(gg[r]);
            float og = 1.f / (1.f + __expf(-go[r]));
            c[r] = fr * c[r] + ir * gr;
            h[r] = og * tanhf(c[r]);
        }
    }
    #pragma unroll
    for (int k = 0; k < H; ++k) hn[(size_t)n * H + k] = h[k];
}

// ---------------- ELU + column-sum readout (deterministic two-stage tree) ----------------
__global__ __launch_bounds__(256)
void elu_colsum_stage1(const float* __restrict__ in, float* __restrict__ part) {
    __shared__ float lds[256];
    const int tid = threadIdx.x;
    const int n = blockIdx.x * 256 + tid;   // grid.x == 16, covers all 4096 nodes
    float v[16];
    #pragma unroll
    for (int f = 0; f < 16; ++f) {
        float x = in[(size_t)n * 16 + f];
        v[f] = x > 0.f ? x : (__expf(x) - 1.f);
    }
    for (int f = 0; f < 16; ++f) {
        lds[tid] = v[f];
        __syncthreads();
        for (int st = 128; st > 0; st >>= 1) {
            if (tid < st) lds[tid] += lds[tid + st];
            __syncthreads();
        }
        if (tid == 0) part[blockIdx.x * 16 + f] = lds[0];
        __syncthreads();
    }
}

__global__ void colsum_stage2(const float* __restrict__ part, float* __restrict__ v) {
    int f = threadIdx.x;
    if (f < 16) {
        float a = 0.f;
        for (int b = 0; b < 16; ++b) a += part[b * 16 + f];
        v[f] = a;
    }
}

// ---------------- final FC: leaky_relu(concat(v1,v2) @ fcW.T + b, 0.1) ----------------
__global__ void final_fc(const float* __restrict__ v, const float* __restrict__ W,
                         const float* __restrict__ b, float* __restrict__ out) {
    int o = threadIdx.x;
    if (o < 16) {
        float z = b[o];
        for (int k = 0; k < 32; ++k) z = fmaf(v[k], W[o * 32 + k], z);
        out[o] = z > 0.f ? z : 0.1f * z;
    }
}

// ==================================================================================
//                                 host orchestration
// ==================================================================================
struct GatScratch {
    unsigned* bits;
    _Float16* hpadT;
    float *sbuf, *tbuf, *mbuf, *cpart, *lpart, *gout;
};

template<int H, int D>
static void run_gat(hipStream_t st, const float* x, int F,
                    const float* W, const float* as, const float* ad, const GatScratch& S) {
    const int HD = H * D;
    gat_transform<<<dim3(NN), dim3(32), 0, st>>>(x, W, as, ad, S.hpadT, S.sbuf, S.tbuf, F, H, D);
    gat_rowmax<H><<<dim3(NN), dim3(256), 0, st>>>(S.bits, S.sbuf, S.tbuf, S.mbuf);
    gat_agg<H, D><<<dim3(NN / 16, H, SPLITJ), dim3(32), 0, st>>>(
        S.bits, S.hpadT, S.sbuf, S.tbuf, S.mbuf, S.cpart, S.lpart);
    combine_norm<<<dim3((NN * HD + 255) / 256), dim3(256), 0, st>>>(
        S.cpart, S.lpart, S.gout, HD, H, D);
}

template<int HD, int JD>
static void run_branch(hipStream_t st, const float* x0, int F,
                       const float* preW, const float* preAs, const float* preAd,
                       const float* jknW, const float* jknAs, const float* jknAd,
                       const float* outW, const float* outAs, const float* outAd,
                       const float* lng, const float* lnb,
                       const float* Wih, const float* Whh, const float* bih, const float* bhh,
                       GatScratch S, float* xa, float* xb, float* seq, float* hn,
                       float* sumpart, float* vdst) {
    const dim3 gN((NN + 127) / 128), bN(128);
    // pre-GAT -> ELU -> LN
    run_gat<1, HD>(st, x0, F, preW, preAs, preAd, S);
    elu_ln<HD><<<gN, bN, 0, st>>>(S.gout, lng, lnb, xa, (float*)nullptr, 0);
    // 6 JKN layers (shared params), collect layer outputs into seq [N, 6, HD]
    float* cur = xa; float* nxt = xb;
    for (int L = 0; L < 6; ++L) {
        run_gat<3, JD>(st, cur, HD, jknW, jknAs, jknAd, S);
        elu_ln<HD><<<gN, bN, 0, st>>>(S.gout, lng, lnb, nxt, seq + L * HD, 6 * HD);
        float* tmp = cur; cur = nxt; nxt = tmp;
    }
    // LSTM over the 6 states -> hn [N, HD]
    lstm_kernel<HD><<<gN, bN, 0, st>>>(seq, Wih, Whh, bih, bhh, hn, 6);
    // out-GAT (H=1, D=16) -> ELU -> column sum
    run_gat<1, 16>(st, hn, HD, outW, outAs, outAd, S);
    elu_colsum_stage1<<<dim3(16), dim3(256), 0, st>>>(S.gout, sumpart);
    colsum_stage2<<<dim3(1), dim3(32), 0, st>>>(sumpart, vdst);
}

extern "C" void kernel_launch(void* const* d_in, const int* in_sizes, int n_in,
                              void* d_out, int out_size, void* d_ws, size_t ws_size,
                              hipStream_t stream) {
    (void)in_sizes; (void)n_in; (void)out_size; (void)ws_size;
    const float* x1   = (const float*)d_in[0];
    const int*   adj1 = (const int*)  d_in[1];
    const float* x2   = (const float*)d_in[2];
    const int*   adj2 = (const int*)  d_in[3];
    const float *pre1_W = (const float*)d_in[4],  *pre1_as = (const float*)d_in[5],  *pre1_ad = (const float*)d_in[6];
    const float *jkn1_W = (const float*)d_in[7],  *jkn1_as = (const float*)d_in[8],  *jkn1_ad = (const float*)d_in[9];
    const float *out1_W = (const float*)d_in[10], *out1_as = (const float*)d_in[11], *out1_ad = (const float*)d_in[12];
    const float *pre2_W = (const float*)d_in[13], *pre2_as = (const float*)d_in[14], *pre2_ad = (const float*)d_in[15];
    const float *jkn2_W = (const float*)d_in[16], *jkn2_as = (const float*)d_in[17], *jkn2_ad = (const float*)d_in[18];
    const float *out2_W = (const float*)d_in[19], *out2_as = (const float*)d_in[20], *out2_ad = (const float*)d_in[21];
    const float *n1_g = (const float*)d_in[22], *n1_b = (const float*)d_in[23];
    const float *n2_g = (const float*)d_in[24], *n2_b = (const float*)d_in[25];
    const float *l1_Wih = (const float*)d_in[26], *l1_Whh = (const float*)d_in[27];
    const float *l1_bih = (const float*)d_in[28], *l1_bhh = (const float*)d_in[29];
    const float *l2_Wih = (const float*)d_in[30], *l2_Whh = (const float*)d_in[31];
    const float *l2_bih = (const float*)d_in[32], *l2_bhh = (const float*)d_in[33];
    const float *fc_W = (const float*)d_in[34], *fc_b = (const float*)d_in[35];

    // ---- workspace carve-out (all buffers written before read) ----
    char* w = (char*)d_ws;
    size_t off = 0;
    auto grab = [&](size_t bytes) -> void* {
        void* p = w + off;
        off = (off + bytes + 255) & ~(size_t)255;
        return p;
    };
    unsigned* bits1 = (unsigned*)grab((size_t)NN * NWORDS * 4);            // 2 MB
    unsigned* bits2 = (unsigned*)grab((size_t)NN * NWORDS * 4);            // 2 MB
    _Float16* hpadT = (_Float16*)grab((size_t)16 * NN * 2);                // 128 KB
    float* xa    = (float*)grab((size_t)NN * 16 * 4);
    float* xb    = (float*)grab((size_t)NN * 16 * 4);
    float* sbuf  = (float*)grab((size_t)NN * 3 * 4);
    float* tbuf  = (float*)grab((size_t)NN * 3 * 4);
    float* mbuf  = (float*)grab((size_t)NN * 3 * 4);
    float* gout  = (float*)grab((size_t)NN * 16 * 4);
    float* seq   = (float*)grab((size_t)NN * 6 * 16 * 4);                  // 1.5 MB
    float* hn    = (float*)grab((size_t)NN * 16 * 4);
    float* cpart = (float*)grab((size_t)SPLITJ * NN * 16 * 4);             // 2 MB
    float* lpart = (float*)grab((size_t)SPLITJ * NN * 3 * 4);              // 384 KB
    float* sumpart = (float*)grab(16 * 16 * 4);
    float* vbuf  = (float*)grab(32 * 4);

    // ---- pack masks once; they stay L2-resident for all 16 GAT layers ----
    pack_mask<<<dim3((NN * NWORDS + 255) / 256), dim3(256), 0, stream>>>(adj1, bits1);
    pack_mask<<<dim3((NN * NWORDS + 255) / 256), dim3(256), 0, stream>>>(adj2, bits2);

    GatScratch S1{bits1, hpadT, sbuf, tbuf, mbuf, cpart, lpart, gout};
    run_branch<12, 4>(stream, x1, 128,
                      pre1_W, pre1_as, pre1_ad, jkn1_W, jkn1_as, jkn1_ad,
                      out1_W, out1_as, out1_ad, n1_g, n1_b,
                      l1_Wih, l1_Whh, l1_bih, l1_bhh,
                      S1, xa, xb, seq, hn, sumpart, vbuf);

    GatScratch S2{bits2, hpadT, sbuf, tbuf, mbuf, cpart, lpart, gout};
    run_branch<6, 2>(stream, x2, 64,
                     pre2_W, pre2_as, pre2_ad, jkn2_W, jkn2_as, jkn2_ad,
                     out2_W, out2_as, out2_ad, n2_g, n2_b,
                     l2_Wih, l2_Whh, l2_bih, l2_bhh,
                     S2, xa, xb, seq, hn, sumpart, vbuf + 16);

    final_fc<<<dim3(1), dim3(32), 0, stream>>>(vbuf, fc_W, fc_b, (float*)d_out);
}
